// MPF_MDR_Model_87076166959612
// MI455X (gfx1250) — matrix operand order repaired and checked by reading, not compiled, not run on hardware
//
#include <hip/hip_runtime.h>
#include <hip/hip_bf16.h>

#define ITEMN   100000
#define PROTON  1024
#define DDIM    128
#define BATCH   4096
#define HISTN   50
#define TOPN    4

typedef _Float16 v8h_t  __attribute__((ext_vector_type(8)));
typedef _Float16 v16h_t __attribute__((ext_vector_type(16)));
typedef float    v8f_t  __attribute__((ext_vector_type(8)));

// ---- WMMA fragment loaders (wave32, V_WMMA_F32_16X16X32_F16) -------------
// A 16x32 f16 (row-major, lda halves): lane row = lane&15,
//   elems 0..7  = K (lane<16 ? 0:8)+0..7 ; elems 8..15 = +16.
__device__ __forceinline__ v16h_t frag_a(const _Float16* base, int lda, int kk, int lane) {
  const int row = lane & 15;
  const int kb  = (lane & 16) ? 8 : 0;
  const _Float16* p = base + row * lda + kk + kb;
  v8h_t lo = *(const v8h_t*)(p);
  v8h_t hi = *(const v8h_t*)(p + 16);
  return __builtin_shufflevector(lo, hi, 0,1,2,3,4,5,6,7,8,9,10,11,12,13,14,15);
}
// B 32x16 f16 stored N-major (baseT[n][k], ldb halves): lane col = lane&15,
//   elems 0..7  = K (lane<16 ? 0:8)+0..7 ; elems 8..15 = +16.
__device__ __forceinline__ v16h_t frag_b(const _Float16* baseT, int ldb, int n0, int kk, int lane) {
  const int col = lane & 15;
  const int kb  = (lane & 16) ? 8 : 0;
  const _Float16* p = baseT + (n0 + col) * ldb + kk + kb;
  v8h_t lo = *(const v8h_t*)(p);
  v8h_t hi = *(const v8h_t*)(p + 16);
  return __builtin_shufflevector(lo, hi, 0,1,2,3,4,5,6,7,8,9,10,11,12,13,14,15);
}
#define WMMA(acc, a, b) \
  __builtin_amdgcn_wmma_f32_16x16x32_f16(false, (a), false, (b), (short)0, (acc), false, false)

// ---- one-shot f16 weight conversion + prototype transpose ----------------
__global__ void k_prep(const float* Wt, const float* Wv, const float* W1,
                       const float* W2, const float* protos,
                       _Float16* Wt_h, _Float16* Wv_h, _Float16* W1_h,
                       _Float16* W2_h, _Float16* protoT_h) {
  const int i = blockIdx.x * blockDim.x + threadIdx.x;
  if (i < DDIM * 384) Wt_h[i] = (_Float16)Wt[i];
  if (i < DDIM * 768) Wv_h[i] = (_Float16)Wv[i];
  if (i < DDIM * 256) W1_h[i] = (_Float16)W1[i];
  if (i < 64 * 128)   W2_h[i] = (_Float16)W2[i];
  if (i < PROTON * DDIM) {                    // protoT[n][k] = protos[k][n]
    const int k = i / DDIM, n = i % DDIM;
    protoT_h[n * PROTON + k] = (_Float16)protos[i];
  }
}

// ---- h_u = (presence mask @ prototypes) / count, WMMA on mask tile -------
__global__ __launch_bounds__(256) void k_hu(const int* inter_items, const int* top_n_index,
                                            const _Float16* protoT_h, float* out_hu) {
  __shared__ __attribute__((aligned(16))) _Float16 maskT[16][PROTON];  // 32 KB
  __shared__ float cnt[16];
  const int b0 = blockIdx.x * 16;
  const int t  = threadIdx.x;

  { v8h_t z = {};
    v8h_t* m8 = (v8h_t*)&maskT[0][0];
    for (int j = t; j < 16 * PROTON / 8; j += 256) m8[j] = z; }
  if (t < 16) cnt[t] = 0.0f;
  __syncthreads();

  const int r = t >> 4, l16 = t & 15, b = b0 + r;
  for (int h = l16; h < HISTN; h += 16) {
    const int item = inter_items[b * HISTN + h];
#pragma unroll
    for (int n = 0; n < TOPN; ++n) {
      maskT[r][top_n_index[item * TOPN + n]] = (_Float16)1.0f;
      maskT[r][top_n_index[(item + ITEMN) * TOPN + n]] = (_Float16)1.0f;
    }
  }
  __syncthreads();
  float partial = 0.0f;
  for (int k = l16 * 64; k < l16 * 64 + 64; ++k) partial += (float)maskT[r][k];
  atomicAdd(&cnt[r], partial);
  __syncthreads();

  const int wv = t >> 5, lane = t & 31, n0 = wv * 16;
  __builtin_prefetch(protoT_h + n0 * PROTON, 0, 3);   // global_prefetch_b8
  v8f_t acc = {};
  for (int kk = 0; kk < PROTON; kk += 32) {
    v16h_t a  = frag_a(&maskT[0][0], PROTON, kk, lane);
    v16h_t bf = frag_b(protoT_h, PROTON, n0, kk, lane);
    acc = WMMA(acc, a, bf);
  }
  const int nc = lane & 15, mbase = (lane & 16) ? 8 : 0;
#pragma unroll
  for (int rr = 0; rr < 8; ++rr) {
    const int m = mbase + rr;
    out_hu[(b0 + m) * DDIM + n0 + nc] = acc[rr] * (1.0f / cnt[m]);
  }
}

// ---- gather pre_text/pre_visual rows, fused PW GEMMs, prompt adds --------
__global__ __launch_bounds__(256) void k_feat(
    const int* nodes_u, const int* nodes_v, const int* nodes_d,
    const float* pre_text, const float* pre_visual, const float* bt, const float* bv,
    const _Float16* Wt_h, const _Float16* Wv_h,
    const float* user_prompt, const float* item_prompt, const float* domain_prompt,
    const float* out_hu, float* out_pu, float* out_pd, _Float16* inter_h) {
  __shared__ __attribute__((aligned(16))) _Float16 At[16][384];  // 12 KB
  __shared__ __attribute__((aligned(16))) _Float16 Av[16][768];  // 24 KB
  const int b0 = blockIdx.x * 16;
  const int t  = threadIdx.x;

  for (int j = t; j < 16 * 384; j += 256) {
    const int r = j / 384, k = j % 384;
    const int v = nodes_v[b0 + r];
    At[r][k] = (_Float16)(pre_text[v * 384 + k] - bt[k]);
  }
  for (int j = t; j < 16 * 768; j += 256) {
    const int r = j / 768, k = j % 768;
    const int v = nodes_v[b0 + r];
    Av[r][k] = (_Float16)(pre_visual[v * 768 + k] - bv[k]);
  }
  for (int j = t; j < 16 * DDIM; j += 256) {     // h_u_hat + prompt outputs
    const int r = j / DDIM, n = j % DDIM, b = b0 + r;
    const float pd = domain_prompt[nodes_d[b] * DDIM + n];
    out_pd[b * DDIM + n] = pd;
    out_pu[b * DDIM + n] = user_prompt[nodes_u[b] * DDIM + n];
    inter_h[b * 256 + n] = (_Float16)(out_hu[b * DDIM + n] + pd);
  }
  __syncthreads();

  const int wv = t >> 5, lane = t & 31, n0 = wv * 16;
  v8f_t acc = {};
  for (int kk = 0; kk < 384; kk += 32) {
    v16h_t a  = frag_a(&At[0][0], 384, kk, lane);
    v16h_t bf = frag_b(Wt_h, 384, n0, kk, lane);
    acc = WMMA(acc, a, bf);
  }
  for (int kk = 0; kk < 768; kk += 32) {
    v16h_t a  = frag_a(&Av[0][0], 768, kk, lane);
    v16h_t bf = frag_b(Wv_h, 768, n0, kk, lane);
    acc = WMMA(acc, a, bf);
  }
  const int nc = lane & 15, mbase = (lane & 16) ? 8 : 0;
#pragma unroll
  for (int rr = 0; rr < 8; ++rr) {
    const int m = mbase + rr, b = b0 + m;
    const float pv = item_prompt[nodes_v[b] * DDIM + n0 + nc];
    inter_h[b * 256 + DDIM + n0 + nc] = (_Float16)(acc[rr] + pv);
  }
}

// ---- f1 = relu(inter @ W1.T + b1) ----------------------------------------
__global__ __launch_bounds__(256) void k_mlp1(const _Float16* inter_h, const _Float16* W1_h,
                                              const float* b1, float* f1) {
  const int b0 = blockIdx.x * 16;
  const int t = threadIdx.x, wv = t >> 5, lane = t & 31, n0 = wv * 16;
  v8f_t acc = {};
  for (int kk = 0; kk < 256; kk += 32) {
    v16h_t a  = frag_a(inter_h + b0 * 256, 256, kk, lane);
    v16h_t bf = frag_b(W1_h, 256, n0, kk, lane);
    acc = WMMA(acc, a, bf);
  }
  const int nc = lane & 15, mbase = (lane & 16) ? 8 : 0;
#pragma unroll
  for (int rr = 0; rr < 8; ++rr) {
    const int m = mbase + rr;
    const float v = acc[rr] + b1[n0 + nc];
    f1[(b0 + m) * 128 + n0 + nc] = v > 0.0f ? v : 0.0f;
  }
}

// ---- training-mode BN stats folded into per-column scale/shift -----------
__global__ __launch_bounds__(256) void k_bnstats(const float* x, const float* g, const float* be,
                                                 int rows, int cols, float* scale, float* shift) {
  __shared__ float ssum[256], ssq[256];
  const int c = blockIdx.x, t = threadIdx.x;
  float s = 0.0f, q = 0.0f;
  for (int r = t; r < rows; r += 256) {
    const float v = x[r * cols + c];
    s += v; q += v * v;
  }
  ssum[t] = s; ssq[t] = q;
  __syncthreads();
  for (int k = 128; k > 0; k >>= 1) {
    if (t < k) { ssum[t] += ssum[t + k]; ssq[t] += ssq[t + k]; }
    __syncthreads();
  }
  if (t == 0) {
    const float m   = ssum[0] / (float)rows;
    const float var = ssq[0] / (float)rows - m * m;
    const float sc  = g[c] * rsqrtf(var + 1e-5f);
    scale[c] = sc;
    shift[c] = be[c] - m * sc;
  }
}

// ---- f2 = relu(bn1(f1) @ W2.T + b2) --------------------------------------
__global__ __launch_bounds__(128) void k_mlp2(const float* f1, const float* scale1,
                                              const float* shift1, const _Float16* W2_h,
                                              const float* b2, float* f2) {
  __shared__ __attribute__((aligned(16))) _Float16 A[16][128];
  const int b0 = blockIdx.x * 16;
  const int t = threadIdx.x;
  for (int j = t; j < 16 * 128; j += 128) {
    const int r = j / 128, k = j % 128;
    A[r][k] = (_Float16)(f1[(b0 + r) * 128 + k] * scale1[k] + shift1[k]);
  }
  __syncthreads();
  const int wv = t >> 5, lane = t & 31, n0 = wv * 16;  // 4 waves x 16 = N 64
  v8f_t acc = {};
  for (int kk = 0; kk < 128; kk += 32) {
    v16h_t a  = frag_a(&A[0][0], 128, kk, lane);
    v16h_t bf = frag_b(W2_h, 128, n0, kk, lane);
    acc = WMMA(acc, a, bf);
  }
  const int nc = lane & 15, mbase = (lane & 16) ? 8 : 0;
#pragma unroll
  for (int rr = 0; rr < 8; ++rr) {
    const int m = mbase + rr;
    const float v = acc[rr] + b2[n0 + nc];
    f2[(b0 + m) * 64 + n0 + nc] = v > 0.0f ? v : 0.0f;
  }
}

// ---- pred = sigmoid(bn2(f2) @ Wp.T + bp) ---------------------------------
__global__ void k_pred(const float* f2, const float* scale2, const float* shift2,
                       const float* Wp, const float* bp, float* pred) {
  const int b = blockIdx.x * blockDim.x + threadIdx.x;
  if (b >= BATCH) return;
  float acc = bp[0];
#pragma unroll 8
  for (int c = 0; c < 64; ++c)
    acc += (f2[b * 64 + c] * scale2[c] + shift2[c]) * Wp[c];
  pred[b] = 1.0f / (1.0f + __expf(-acc));
}

extern "C" void kernel_launch(void* const* d_in, const int* in_sizes, int n_in,
                              void* d_out, int out_size, void* d_ws, size_t ws_size,
                              hipStream_t stream) {
  (void)in_sizes; (void)n_in; (void)out_size; (void)ws_size;
  const int*   nodes_u       = (const int*)  d_in[0];
  const int*   nodes_v       = (const int*)  d_in[1];
  const int*   nodes_d       = (const int*)  d_in[2];
  const int*   inter_items   = (const int*)  d_in[3];
  const int*   top_n_index   = (const int*)  d_in[4];
  const float* prototypes    = (const float*)d_in[5];
  const float* pre_text      = (const float*)d_in[6];
  const float* pre_visual    = (const float*)d_in[7];
  const float* Wt            = (const float*)d_in[8];
  const float* bt            = (const float*)d_in[9];
  const float* Wv            = (const float*)d_in[10];
  const float* bv            = (const float*)d_in[11];
  const float* user_prompt   = (const float*)d_in[12];
  const float* item_prompt   = (const float*)d_in[13];
  const float* domain_prompt = (const float*)d_in[14];
  const float* W1            = (const float*)d_in[15];
  const float* b1            = (const float*)d_in[16];
  const float* g1            = (const float*)d_in[17];
  const float* be1           = (const float*)d_in[18];
  const float* W2            = (const float*)d_in[19];
  const float* b2            = (const float*)d_in[20];
  const float* g2            = (const float*)d_in[21];
  const float* be2           = (const float*)d_in[22];
  const float* Wp            = (const float*)d_in[23];
  const float* bp            = (const float*)d_in[24];

  float* out    = (float*)d_out;                 // (pred, p_u, h_u, p_d) flat
  float* pred   = out;
  float* out_pu = out + BATCH;
  float* out_hu = out + BATCH + BATCH * DDIM;
  float* out_pd = out + BATCH + 2 * BATCH * DDIM;

  char* w = (char*)d_ws;
  _Float16* Wt_h     = (_Float16*)w; w += DDIM * 384 * 2;
  _Float16* Wv_h     = (_Float16*)w; w += DDIM * 768 * 2;
  _Float16* W1_h     = (_Float16*)w; w += DDIM * 256 * 2;
  _Float16* W2_h     = (_Float16*)w; w += 64 * 128 * 2;
  _Float16* protoT_h = (_Float16*)w; w += PROTON * DDIM * 2;
  _Float16* inter_h  = (_Float16*)w; w += BATCH * 256 * 2;
  float*    f1       = (float*)w;    w += BATCH * 128 * 4;
  float*    f2       = (float*)w;    w += BATCH * 64 * 4;
  float*    scale1   = (float*)w;    w += 128 * 4;
  float*    shift1   = (float*)w;    w += 128 * 4;
  float*    scale2   = (float*)w;    w += 64 * 4;
  float*    shift2   = (float*)w;    w += 64 * 4;

  k_prep<<<512, 256, 0, stream>>>(Wt, Wv, W1, W2, prototypes,
                                  Wt_h, Wv_h, W1_h, W2_h, protoT_h);
  k_hu<<<BATCH / 16, 256, 0, stream>>>(inter_items, top_n_index, protoT_h, out_hu);
  k_feat<<<BATCH / 16, 256, 0, stream>>>(nodes_u, nodes_v, nodes_d,
                                         pre_text, pre_visual, bt, bv, Wt_h, Wv_h,
                                         user_prompt, item_prompt, domain_prompt,
                                         out_hu, out_pu, out_pd, inter_h);
  k_mlp1<<<BATCH / 16, 256, 0, stream>>>(inter_h, W1_h, b1, f1);
  k_bnstats<<<128, 256, 0, stream>>>(f1, g1, be1, BATCH, 128, scale1, shift1);
  k_mlp2<<<BATCH / 16, 128, 0, stream>>>(f1, scale1, shift1, W2_h, b2, f2);
  k_bnstats<<<64, 256, 0, stream>>>(f2, g2, be2, BATCH, 64, scale2, shift2);
  k_pred<<<BATCH / 256, 256, 0, stream>>>(f2, scale2, shift2, Wp, bp, pred);
}
